// LeakyIntegrateFireAttention_77369540870292
// MI455X (gfx1250) — matrix operand order repaired and checked by reading, not compiled, not run on hardware
//
#include <hip/hip_runtime.h>
#include <hip/hip_bf16.h>
#include <math.h>

// ---------------------------------------------------------------------------
// LIF attention for MI455X (gfx1250, wave32, WMMA).
// All GEMM-shaped math runs on v_wmma_f32_16x16x32_f16 (f16 in, f32 acc).
// ---------------------------------------------------------------------------

typedef __attribute__((ext_vector_type(16))) _Float16 v16h;
typedef __attribute__((ext_vector_type(8)))  _Float16 v8h;
typedef __attribute__((ext_vector_type(8)))  float    v8f;

#define NB 4
#define NT 1024
#define NC 1024
#define NH 16
#define HS 64
#define MASKV (-1e30f)

__device__ __forceinline__ int lane_id() { return (int)(threadIdx.x & 31); }

// A operand, 16(M) x 32(K), row-major source, ld in elements.
// ISA layout: lanes 0-15 -> M=0..15 hold K 0..7 (v0-3) and 16..23 (v4-7);
// lanes 16-31 -> same M hold K 8..15 and 24..31.
__device__ __forceinline__ v16h load_frag_a(const _Float16* p, int ld) {
  const int l  = lane_id();
  const int m  = l & 15;
  const int hi = l >> 4;
  const _Float16* row = p + (size_t)m * ld;
  v8h lo = *(const v8h*)(row + hi * 8);
  v8h hh = *(const v8h*)(row + 16 + hi * 8);
  v16h r;
#pragma unroll
  for (int i = 0; i < 8; ++i) { r[i] = lo[i]; r[8 + i] = hh[i]; }
  return r;
}

// B operand, 32(K) x 16(N), row-major [K][N] source (lane%16 = N index;
// lanes 0-15 hold K 0..7 and 16..23, lanes 16-31 hold K 8..15 and 24..31).
__device__ __forceinline__ v16h load_frag_b(const _Float16* p, int ld) {
  const int l = lane_id();
  const int n  = l & 15;
  const int hi = l >> 4;
  const _Float16* col = p + n;
  v16h r;
#pragma unroll
  for (int i = 0; i < 8; ++i) {
    r[i] = col[(size_t)(hi * 8 + i) * ld];
    r[8 + i] = col[(size_t)(16 + hi * 8 + i) * ld];
  }
  return r;
}

__device__ __forceinline__ v8f wmma16(v16h a, v16h b, v8f c) {
  return __builtin_amdgcn_wmma_f32_16x16x32_f16(false, a, false, b,
                                                (short)0, c, false, false);
}

__device__ __forceinline__ float sigm(float x) { return 1.f / (1.f + __expf(-x)); }
__device__ __forceinline__ float softp(float x) { return log1pf(__expf(x)); }

// ------------------------------- utility kernels ---------------------------

__global__ void cast_f32_f16_kernel(const float* __restrict__ s,
                                    _Float16* __restrict__ d, int n) {
  int i = blockIdx.x * blockDim.x + threadIdx.x;
  if (i < n) d[i] = (_Float16)s[i];
}

__global__ void zero_f32_kernel(float* __restrict__ p, int n) {
  int i = blockIdx.x * blockDim.x + threadIdx.x;
  if (i < n) p[i] = 0.f;
}

// ------------------------------- GEMM --------------------------------------
// C[M,N] = A[M,K](f16,row-major) * B[K,N](f16,row-major) + bias.
// 256 threads = 8 waves; block tile 128x128; wave tile 64(M)x32(N).
// MODE 0: qkv epilogue (scatter q*0.125 / kT / v). MODE 1: fp32 out.
template <int MODE>
__global__ void __launch_bounds__(256)
gemm_f16_kernel(const _Float16* __restrict__ A, const _Float16* __restrict__ Bm,
                const float* __restrict__ bias,
                _Float16* __restrict__ q_out, _Float16* __restrict__ kT_out,
                _Float16* __restrict__ v_out, float* __restrict__ f_out,
                int M, int N, int K) {
  const int wid = threadIdx.x >> 5;
  const int wm  = wid & 1;   // 2 waves along M
  const int wn  = wid >> 1;  // 4 waves along N
  const int m_base = blockIdx.y * 128 + wm * 64;
  const int n_base = blockIdx.x * 128 + wn * 32;

  v8f zero = {};
  v8f acc[4][2];
#pragma unroll
  for (int i = 0; i < 4; ++i)
#pragma unroll
    for (int j = 0; j < 2; ++j) acc[i][j] = zero;

  for (int k0 = 0; k0 < K; k0 += 32) {
    v16h af[4];
#pragma unroll
    for (int i = 0; i < 4; ++i)
      af[i] = load_frag_a(A + (size_t)(m_base + i * 16) * K + k0, K);
    v16h bf[2];
#pragma unroll
    for (int j = 0; j < 2; ++j)
      bf[j] = load_frag_b(Bm + (size_t)k0 * N + n_base + j * 16, N);
#pragma unroll
    for (int i = 0; i < 4; ++i)
#pragma unroll
      for (int j = 0; j < 2; ++j) acc[i][j] = wmma16(af[i], bf[j], acc[i][j]);
  }

  const int l  = lane_id();
  const int ml = (l >> 4) * 8;  // C/D layout: lanes 16-31 hold M=8..15
  const int nl = l & 15;
#pragma unroll
  for (int i = 0; i < 4; ++i) {
#pragma unroll
    for (int j = 0; j < 2; ++j) {
#pragma unroll
      for (int r = 0; r < 8; ++r) {
        const int m = m_base + i * 16 + ml + r;
        const int n = n_base + j * 16 + nl;
        const float v = acc[i][j][r] + bias[n];
        if (MODE == 0) {
          const int b = m >> 10, t = m & (NT - 1);
          if (n < NC) {  // Q, pre-scaled by 1/sqrt(hs)
            const int h = n >> 6, d2 = n & 63;
            q_out[(((size_t)(b * NH + h) * NT + t) << 6) + d2] =
                (_Float16)(v * 0.125f);
          } else if (n < 2 * NC) {  // K transposed -> [b,h,hs,T]
            const int c = n - NC;
            const int h = c >> 6, d2 = c & 63;
            kT_out[((size_t)(b * NH + h) * HS + d2) * NT + t] = (_Float16)v;
          } else {  // V -> [b,h,T,hs]
            const int c = n - 2 * NC;
            const int h = c >> 6, d2 = c & 63;
            v_out[(((size_t)(b * NH + h) * NT + t) << 6) + d2] = (_Float16)v;
          }
        } else {
          f_out[(size_t)m * N + n] = v;
        }
      }
    }
  }
}

// --------------------------- attention pass 1 ------------------------------
// Per block: one (b,h) and 128 query rows; wave -> 16 query rows.
// Loop 1: online softmax stats (row max m, row sum l) over causal key tiles.
// Loop 2: recompute P, atomically accumulate column sums (for column_load).
__global__ void __launch_bounds__(256)
attn_pass1_kernel(const _Float16* __restrict__ qbuf,
                  const _Float16* __restrict__ kT,
                  float* __restrict__ mbuf, float* __restrict__ lbuf,
                  float* __restrict__ colsum) {
  const int qt_per = NT / 128;
  const int bh = blockIdx.x / qt_per;
  const int qt = blockIdx.x % qt_per;
  const int wid = threadIdx.x >> 5;
  const int q0 = qt * 128 + wid * 16;
  const int l  = lane_id();
  const int ml = (l >> 4) * 8;
  const int nl = l & 15;

  const _Float16* Q  = qbuf + ((size_t)bh * NT + q0) * HS;
  const _Float16* Kt = kT + (size_t)bh * HS * NT;
  const v16h aq0 = load_frag_a(Q, HS);
  const v16h aq1 = load_frag_a(Q + 32, HS);
  const v8f zero = {};
  const int ntiles = (q0 >> 4) + 1;

  float mx[8], sm[8];
#pragma unroll
  for (int r = 0; r < 8; ++r) { mx[r] = MASKV; sm[r] = 0.f; }

  for (int t = 0; t < ntiles; ++t) {
    const int kt0 = t << 4;
    v8f s = zero;
    s = wmma16(aq0, load_frag_b(Kt + kt0, NT), s);
    s = wmma16(aq1, load_frag_b(Kt + 32 * NT + kt0, NT), s);
#pragma unroll
    for (int r = 0; r < 8; ++r) {
      const float v = (kt0 + nl <= q0 + ml + r) ? s[r] : MASKV;
      const float nm = fmaxf(mx[r], v);
      sm[r] = sm[r] * __expf(mx[r] - nm) + __expf(v - nm);
      mx[r] = nm;
    }
  }
  // merge lane-partial (max,sum) across the 16 lanes of each half
#pragma unroll
  for (int d = 1; d < 16; d <<= 1) {
#pragma unroll
    for (int r = 0; r < 8; ++r) {
      const float om = __shfl_xor(mx[r], d, 32);
      const float os = __shfl_xor(sm[r], d, 32);
      const float nm = fmaxf(mx[r], om);
      sm[r] = sm[r] * __expf(mx[r] - nm) + os * __expf(om - nm);
      mx[r] = nm;
    }
  }
  if (nl == 0) {
#pragma unroll
    for (int r = 0; r < 8; ++r) {
      mbuf[(size_t)bh * NT + q0 + ml + r] = mx[r];
      lbuf[(size_t)bh * NT + q0 + ml + r] = sm[r];
    }
  }
  float linv[8];
#pragma unroll
  for (int r = 0; r < 8; ++r) linv[r] = 1.f / sm[r];

  for (int t = 0; t < ntiles; ++t) {
    const int kt0 = t << 4;
    v8f s = zero;
    s = wmma16(aq0, load_frag_b(Kt + kt0, NT), s);
    s = wmma16(aq1, load_frag_b(Kt + 32 * NT + kt0, NT), s);
    float cs = 0.f;
#pragma unroll
    for (int r = 0; r < 8; ++r) {
      const float v = (kt0 + nl <= q0 + ml + r) ? s[r] : MASKV;
      cs += __expf(v - mx[r]) * linv[r];
    }
    cs += __shfl_xor(cs, 16, 32);  // combine M 0-7 with M 8-15 halves
    if (l < 16) atomicAdd(&colsum[(size_t)bh * NT + kt0 + l], cs);
  }
}

// --------------------------- attention pass 2 ------------------------------
// Loop 1: recompute P, apply LIF gate, get per-row sum of mod (normalizer).
// Loop 2: recompute P per key-tile pair, build normalized mod tile in LDS
//         (S-layout -> A-fragment relayout), WMMA against V, accumulate
//         column sums of mod for new_refractory.
__global__ void __launch_bounds__(256)
attn_pass2_kernel(const _Float16* __restrict__ qbuf,
                  const _Float16* __restrict__ kT,
                  const _Float16* __restrict__ vbuf,
                  const float* __restrict__ mbuf, const float* __restrict__ lbuf,
                  const float* __restrict__ colsum,
                  const float* __restrict__ refr,
                  const float* __restrict__ p_thr, const float* __restrict__ p_leak,
                  const float* __restrict__ p_steep, const float* __restrict__ p_rs,
                  const float* __restrict__ p_cw,
                  _Float16* __restrict__ yh, float* __restrict__ cmod) {
  __shared__ __align__(16) _Float16 ldsP[8 * 16 * 32];

  const int qt_per = NT / 128;
  const int bh = blockIdx.x / qt_per;
  const int qt = blockIdx.x % qt_per;
  const int wid = threadIdx.x >> 5;
  const int q0 = qt * 128 + wid * 16;
  const int l  = lane_id();
  const int ml = (l >> 4) * 8;
  const int nl = l & 15;
  const int h = bh % NH;
  const int b = bh / NH;

  const float thr = fabsf(p_thr[h]) * 0.1f;
  const float lk  = sigm(p_leak[h]);
  const float st  = softp(p_steep[h]);
  const float rs  = softp(p_rs[h]);
  const float cw  = sigm(p_cw[h]);
  const float invT = 1.f / (float)NT;

  const _Float16* Q  = qbuf + ((size_t)bh * NT + q0) * HS;
  const _Float16* Kt = kT + (size_t)bh * HS * NT;
  const _Float16* V  = vbuf + (size_t)bh * NT * HS;
  const v16h aq0 = load_frag_a(Q, HS);
  const v16h aq1 = load_frag_a(Q + 32, HS);
  const v8f zero = {};
  const int ntiles = (q0 >> 4) + 1;

  float mx[8], linv[8];
#pragma unroll
  for (int r = 0; r < 8; ++r) {
    mx[r]   = mbuf[(size_t)bh * NT + q0 + ml + r];
    linv[r] = 1.f / lbuf[(size_t)bh * NT + q0 + ml + r];
  }

  // loop 1: row sums of un-normalized mod
  float rs2[8];
#pragma unroll
  for (int r = 0; r < 8; ++r) rs2[r] = 0.f;
  for (int t = 0; t < ntiles; ++t) {
    const int kt0 = t << 4;
    v8f s = zero;
    s = wmma16(aq0, load_frag_b(Kt + kt0, NT), s);
    s = wmma16(aq1, load_frag_b(Kt + 32 * NT + kt0, NT), s);
    const int key = kt0 + nl;
    const float et = thr + rs * (colsum[(size_t)bh * NT + key] * invT) +
                     cw * refr[(size_t)b * NT + key];
#pragma unroll
    for (int r = 0; r < 8; ++r) {
      const float v = (key <= q0 + ml + r) ? s[r] : MASKV;
      const float p = __expf(v - mx[r]) * linv[r];
      const float fire = sigm(st * (p - et));
      rs2[r] += p * (fire + lk * (1.f - fire));
    }
  }
#pragma unroll
  for (int d = 1; d < 16; d <<= 1)
#pragma unroll
    for (int r = 0; r < 8; ++r) rs2[r] += __shfl_xor(rs2[r], d, 32);
  float rnorm[8];
#pragma unroll
  for (int r = 0; r < 8; ++r) rnorm[r] = 1.f / (rs2[r] + 1e-8f);

  // loop 2: PV with normalized mod + column sums of mod
  v8f accy[4];
#pragma unroll
  for (int dt = 0; dt < 4; ++dt) accy[dt] = zero;
  _Float16* myP = ldsP + wid * (16 * 32);
  const int npairs = (ntiles + 1) >> 1;

  for (int kp = 0; kp < npairs; ++kp) {
#pragma unroll
    for (int j = 0; j < 2; ++j) {
      const int t = kp * 2 + j;
      const int kt0 = t << 4;
      float mu[8];
      if (t < ntiles) {
        v8f s = zero;
        s = wmma16(aq0, load_frag_b(Kt + kt0, NT), s);
        s = wmma16(aq1, load_frag_b(Kt + 32 * NT + kt0, NT), s);
        const int key = kt0 + nl;
        const float et = thr + rs * (colsum[(size_t)bh * NT + key] * invT) +
                         cw * refr[(size_t)b * NT + key];
        float cs = 0.f;
#pragma unroll
        for (int r = 0; r < 8; ++r) {
          const float v = (key <= q0 + ml + r) ? s[r] : MASKV;
          const float p = __expf(v - mx[r]) * linv[r];
          const float fire = sigm(st * (p - et));
          mu[r] = p * (fire + lk * (1.f - fire)) * rnorm[r];
          cs += mu[r];
        }
        cs += __shfl_xor(cs, 16, 32);
        if (l < 16) atomicAdd(&cmod[(size_t)bh * NT + kt0 + l], cs);
      } else {
#pragma unroll
        for (int r = 0; r < 8; ++r) mu[r] = 0.f;
      }
#pragma unroll
      for (int r = 0; r < 8; ++r)
        myP[(ml + r) * 32 + j * 16 + nl] = (_Float16)mu[r];
    }
    __builtin_amdgcn_wave_barrier();  // keep LDS store->load order (same wave)
    const v16h ap = load_frag_a(myP, 32);
    __builtin_amdgcn_wave_barrier();
#pragma unroll
    for (int dt = 0; dt < 4; ++dt) {
      const v16h bv = load_frag_b(V + (size_t)(kp * 32) * HS + dt * 16, HS);
      accy[dt] = wmma16(ap, bv, accy[dt]);
    }
  }

  // write per-head output into [B*T, C] f16 for the projection GEMM
#pragma unroll
  for (int dt = 0; dt < 4; ++dt) {
#pragma unroll
    for (int r = 0; r < 8; ++r) {
      const int m = q0 + ml + r;
      const int d2 = dt * 16 + nl;
      yh[((size_t)b * NT + m) * NC + h * HS + d2] = (_Float16)accy[dt][r];
    }
  }
}

// new_refractory[b,k] = sum_h cmod[b,h,k] / (H*T)
__global__ void refr_kernel(const float* __restrict__ cmod,
                            float* __restrict__ outr) {
  int i = blockIdx.x * blockDim.x + threadIdx.x;
  if (i < NB * NT) {
    const int b = i >> 10, k = i & (NT - 1);
    float s = 0.f;
#pragma unroll
    for (int h = 0; h < NH; ++h) s += cmod[(size_t)(b * NH + h) * NT + k];
    outr[i] = s * (1.f / (float)(NH * NT));
  }
}

// ---------------------------------------------------------------------------

extern "C" void kernel_launch(void* const* d_in, const int* in_sizes, int n_in,
                              void* d_out, int out_size, void* d_ws,
                              size_t ws_size, hipStream_t stream) {
  (void)in_sizes; (void)n_in; (void)out_size; (void)ws_size;

  const float* x      = (const float*)d_in[0];
  const float* refr   = (const float*)d_in[1];
  const float* W_attn = (const float*)d_in[2];
  const float* b_attn = (const float*)d_in[3];
  const float* W_proj = (const float*)d_in[4];
  const float* b_proj = (const float*)d_in[5];
  const float* p_thr  = (const float*)d_in[6];
  const float* p_leak = (const float*)d_in[7];
  const float* p_st   = (const float*)d_in[8];
  const float* p_rs   = (const float*)d_in[9];
  const float* p_cw   = (const float*)d_in[10];

  float* y_out = (float*)d_out;                  // [B,T,C]
  float* r_out = y_out + (size_t)NB * NT * NC;   // [B,T]

  char* w = (char*)d_ws;
  size_t off = 0;
  auto alloc = [&](size_t bytes) {
    void* p = w + off;
    off += (bytes + 255) & ~(size_t)255;
    return p;
  };
  _Float16* xh  = (_Float16*)alloc((size_t)NB * NT * NC * 2);          // 8 MB
  _Float16* Wah = (_Float16*)alloc((size_t)NC * 3 * NC * 2);           // 6 MB
  _Float16* Wph = (_Float16*)alloc((size_t)NC * NC * 2);               // 2 MB
  _Float16* qb  = (_Float16*)alloc((size_t)NB * NH * NT * HS * 2);     // 8 MB
  _Float16* kT  = (_Float16*)alloc((size_t)NB * NH * NT * HS * 2);     // 8 MB
  _Float16* vb  = (_Float16*)alloc((size_t)NB * NH * NT * HS * 2);     // 8 MB
  _Float16* yh  = (_Float16*)alloc((size_t)NB * NT * NC * 2);          // 8 MB
  float* mbuf   = (float*)alloc((size_t)NB * NH * NT * 4);
  float* lbuf   = (float*)alloc((size_t)NB * NH * NT * 4);
  float* colsum = (float*)alloc((size_t)NB * NH * NT * 4);
  float* cmod   = (float*)alloc((size_t)NB * NH * NT * 4);

  const int n_x = NB * NT * NC;
  const int n_wa = NC * 3 * NC;
  const int n_wp = NC * NC;
  const int n_bh = NB * NH * NT;

  cast_f32_f16_kernel<<<(n_x + 255) / 256, 256, 0, stream>>>(x, xh, n_x);
  cast_f32_f16_kernel<<<(n_wa + 255) / 256, 256, 0, stream>>>(W_attn, Wah, n_wa);
  cast_f32_f16_kernel<<<(n_wp + 255) / 256, 256, 0, stream>>>(W_proj, Wph, n_wp);
  zero_f32_kernel<<<(n_bh + 255) / 256, 256, 0, stream>>>(colsum, n_bh);
  zero_f32_kernel<<<(n_bh + 255) / 256, 256, 0, stream>>>(cmod, n_bh);

  // qkv = x @ W_attn + b ; scatter into q (scaled), kT, v
  gemm_f16_kernel<0><<<dim3(3 * NC / 128, (NB * NT) / 128), 256, 0, stream>>>(
      xh, Wah, b_attn, qb, kT, vb, nullptr, NB * NT, 3 * NC, NC);

  const int attn_blocks = NB * NH * (NT / 128);
  attn_pass1_kernel<<<attn_blocks, 256, 0, stream>>>(qb, kT, mbuf, lbuf, colsum);
  attn_pass2_kernel<<<attn_blocks, 256, 0, stream>>>(
      qb, kT, vb, mbuf, lbuf, colsum, refr, p_thr, p_leak, p_st, p_rs, p_cw,
      yh, cmod);
  refr_kernel<<<(NB * NT + 255) / 256, 256, 0, stream>>>(cmod, r_out);

  // y = yh @ W_proj + b
  gemm_f16_kernel<1><<<dim3(NC / 128, (NB * NT) / 128), 256, 0, stream>>>(
      yh, Wph, b_proj, nullptr, nullptr, nullptr, y_out, NB * NT, NC, NC);
}
